// IlluminationLayer_19370302505465
// MI455X (gfx1250) — matrix operand order repaired and checked by reading, not compiled, not run on hardware
//
#include <hip/hip_runtime.h>
#include <hip/hip_bf16.h>

typedef __attribute__((ext_vector_type(16))) _Float16 v16h;
typedef __attribute__((ext_vector_type(8)))  _Float16 v8h;
typedef __attribute__((ext_vector_type(8)))  float    v8f;

#define NT     2048
#define KOUT   8
#define NBATCH 16384

// Workspace layout (on d_ws, as float*):
//   w[0]          : inpsum = sum(relu(illumination))
//   w[1..8]       : rowsum[k] = sum_n cmat[k,n]
//   gh  = (_Float16*)(w+16)        : g[n] circular conv, f16         (4 KB)
//   ch  = gh + NT                  : cmat in f16, row-major [8,2048]  (32 KB)
//   csh = (float*)(ch + KOUT*NT)   : Csh[16,2048] f32 (rows 8..15 pad)(128 KB)

// ---------------------------------------------------------------------------
// Kernel 1: setup. inpsum, rowsums, g = circ_conv(relu(ill), irf), f16 packs.
// ---------------------------------------------------------------------------
__global__ void __launch_bounds__(256)
setup_kernel(const float* __restrict__ ill, const float* __restrict__ irf,
             const float* __restrict__ cmat, float* __restrict__ w) {
  __shared__ float s_inp[NT];
  __shared__ float s_irf[NT];
  __shared__ float red[256];
  const int tid = threadIdx.x;

  float psum = 0.f;
  for (int i = tid; i < NT; i += 256) {
    float v = ill[i];
    v = v > 0.f ? v : 0.f;           // relu
    s_inp[i] = v;
    psum += v;
    s_irf[i] = irf[i];
  }
  red[tid] = psum;
  __syncthreads();
  for (int s = 128; s > 0; s >>= 1) {
    if (tid < s) red[tid] += red[tid + s];
    __syncthreads();
  }
  if (tid == 0) w[0] = red[0];

  _Float16* gh = (_Float16*)(w + 16);
  _Float16* ch = gh + NT;

  // g[n] = sum_m inp[m] * irf[(n-m) mod N]  (== irfft(rfft(x)*rfft(irf)))
  for (int n = tid; n < NT; n += 256) {
    float acc = 0.f;
    for (int m = 0; m < NT; ++m)
      acc += s_inp[m] * s_irf[(n - m) & (NT - 1)];
    gh[n] = (_Float16)acc;
  }

  // cmat -> f16
  for (int i = tid; i < KOUT * NT; i += 256)
    ch[i] = (_Float16)cmat[i];

  // rowsum[k]
  for (int k = 0; k < KOUT; ++k) {
    float p = 0.f;
    for (int n = tid; n < NT; n += 256) p += cmat[k * NT + n];
    __syncthreads();
    red[tid] = p;
    __syncthreads();
    for (int s = 128; s > 0; s >>= 1) {
      if (tid < s) red[tid] += red[tid + s];
      __syncthreads();
    }
    if (tid == 0) w[1 + k] = red[0];
  }
}

// ---------------------------------------------------------------------------
// Kernel 2: Csh[k,s] = sum_n cmat[k,n] * g[(n-s) mod N]  via WMMA f16->f32.
// A = cmat zero-padded to 16x2048 in LDS (unconditional b128 loads).
// B[kk,s] = g[(n+kk-s) & 2047] gathered from a DOUBLED g copy in LDS so the
// 16-element window never wraps: one address + immediate offsets per iter.
// One wave32 per 16-column s-tile; 32 blocks x 4 waves = 128 tiles.
// ---------------------------------------------------------------------------
__global__ void __launch_bounds__(128)
corr_wmma_kernel(float* __restrict__ w) {
  __shared__ _Float16 sgd[2 * NT];    // doubled g: 8 KB
  __shared__ _Float16 sc[16 * NT];    // cmat padded to 16 rows: 64 KB
  _Float16* gh  = (_Float16*)(w + 16);
  _Float16* ch  = gh + NT;
  float*    csh = (float*)(ch + KOUT * NT);

  const int tid = threadIdx.x;
  for (int i = tid; i < NT; i += 128) {
    _Float16 v = gh[i];
    sgd[i]      = v;
    sgd[i + NT] = v;
  }
  for (int i = tid; i < KOUT * NT; i += 128) sc[i] = ch[i];
  for (int i = KOUT * NT + tid; i < 16 * NT; i += 128) sc[i] = (_Float16)0.f;
  __syncthreads();

  const int wave = tid >> 5;
  const int lane = tid & 31;
  const int s0   = (blockIdx.x * 4 + wave) * 16;   // this wave's s-tile base

  const int m       = lane & 15;   // A row (cmat k, padded) / B+C column
  const int hi      = lane >> 4;   // half-wave selector
  const int a_kbase = hi * 16;     // A: elem j -> K = a_kbase + j
  const int b_kbase = hi * 16;     // B: elem j -> row K = b_kbase + j of column m
  const int tbase   = b_kbase - s0 - m + 2 * NT;   // per-lane window base (in [2049, 4112])

  v8f c = {};
  for (int n = 0; n < NT; n += 32) {
    // A fragment: two contiguous, 16B-aligned 8-half runs -> ds_load_b128 x2
    const v8h* pa = (const v8h*)(&sc[m * NT + n + a_kbase]);
    v8h alo = pa[0];       // K = a_kbase .. a_kbase+7
    v8h ahi = pa[1];       // K = a_kbase+8 .. a_kbase+15
    v16h a = __builtin_shufflevector(alo, ahi,
                                     0, 1, 2, 3, 4, 5, 6, 7,
                                     8, 9, 10, 11, 12, 13, 14, 15);
    // B fragment: 16-half circulant window, no wrap thanks to doubled g
    const int t = n + tbase;       // t in [2049, 6128]; t+15 <= 6143
    v16h b;
#pragma unroll
    for (int j = 0; j < 16; ++j) b[j] = sgd[t + j];

    // D = A x B + C ; emits v_wmma_f32_16x16x32_f16
    c = __builtin_amdgcn_wmma_f32_16x16x32_f16(false, a, false, b,
                                               (short)0, c, false, false);
  }

  // C/D layout: VGPR r -> row (hi*8 + r), column (s0 + m)
#pragma unroll
  for (int r = 0; r < 8; ++r) {
    const int row = hi * 8 + r;
    csh[row * NT + s0 + m] = c[r];
  }
}

// ---------------------------------------------------------------------------
// Kernel 3: out[b,k] = scale_b * Csh[k, shift_b] + amb_b * rowsum[k]
// ---------------------------------------------------------------------------
__global__ void __launch_bounds__(256)
finalize_kernel(const int* __restrict__ bins, const float* __restrict__ pc,
                const float* __restrict__ sbr, const float* __restrict__ w,
                float* __restrict__ out) {
  const int b = blockIdx.x * blockDim.x + threadIdx.x;
  if (b >= NBATCH) return;
  const _Float16* gh  = (const _Float16*)(w + 16);
  const float*    csh = (const float*)(gh + NT + KOUT * NT);

  const float inpsum = w[0];
  const int   shift  = ((bins[b] % NT) + NT) & (NT - 1);
  const float p      = pc[b];
  const float scale  = p / inpsum;
  const float amb    = (p / sbr[b]) * (1.0f / NT);
#pragma unroll
  for (int k = 0; k < KOUT; ++k)
    out[b * KOUT + k] = scale * csh[k * NT + shift] + amb * w[1 + k];
}

extern "C" void kernel_launch(void* const* d_in, const int* in_sizes, int n_in,
                              void* d_out, int out_size, void* d_ws, size_t ws_size,
                              hipStream_t stream) {
  const int*   bins = (const int*)d_in[0];
  const float* pc   = (const float*)d_in[1];
  const float* sbr  = (const float*)d_in[2];
  const float* ill  = (const float*)d_in[3];
  const float* irf  = (const float*)d_in[4];
  const float* cmat = (const float*)d_in[5];
  float* w   = (float*)d_ws;
  float* out = (float*)d_out;

  setup_kernel<<<1, 256, 0, stream>>>(ill, irf, cmat, w);
  corr_wmma_kernel<<<32, 128, 0, stream>>>(w);
  finalize_kernel<<<(NBATCH + 255) / 256, 256, 0, stream>>>(bins, pc, sbr, w, out);
}
